// ShallowLSTMNet_48198122996285
// MI455X (gfx1250) — matrix operand order repaired and checked by reading, not compiled, not run on hardware
//
#include <hip/hip_runtime.h>
#include <hip/hip_bf16.h>

typedef __attribute__((ext_vector_type(16))) _Float16 v16h;
typedef __attribute__((ext_vector_type(8)))  _Float16 v8h;
typedef __attribute__((ext_vector_type(8)))  float    v8f;

#define T_LEN 1000
#define HID   22           // lstm hidden
#define NGATE 88           // 4*HID
#define CCH   22           // input channels C
#define BATCH 64
#define KDIM  484          // C*H contraction dim of the conv GEMM
#define KPAD  512          // padded to 16 WMMA k-steps of 32
#define NSEQ  (BATCH*CCH)  // 1408 independent scalar LSTM sequences

// fast activations on the VALU transcendental pipe (v_exp_f32 / v_rcp_f32)
__device__ __forceinline__ float sigmoid_fast(float x) {
    return __builtin_amdgcn_rcpf(1.0f + __expf(-x));
}
__device__ __forceinline__ float tanh_fast(float x) {
    return 1.0f - 2.0f * __builtin_amdgcn_rcpf(__expf(2.0f * x) + 1.0f);
}

// ---------------------------------------------------------------------------
// Kernel 0: zero the K-padding (kk = 484..511) of every hs row so the conv
// GEMM can load unguarded b128 A-fragments.
// ---------------------------------------------------------------------------
__global__ __launch_bounds__(256)
void pad_zero_kernel(_Float16* __restrict__ hsbuf)   // [B*T, 512] f16
{
    const size_t row = (size_t)blockIdx.x * 256 + threadIdx.x;
    if (row < (size_t)BATCH * T_LEN) {
        unsigned long long* p =
            (unsigned long long*)((char*)hsbuf + row * (KPAD * 2) + KDIM * 2);
        #pragma unroll
        for (int i = 0; i < 7; ++i) p[i] = 0ull;     // 28 halves = 56 bytes
    }
}

// ---------------------------------------------------------------------------
// Kernel 1: batched LSTM recurrence via WMMA.
// 128 threads (4 waves) own 16 contiguous sequences.
//   gates[16,88] = xproj[16,88] + h[16,22] @ Whh^T[22,88]
// M=16, K=22 (pad 32), N=88 (pad 96 -> 6 tiles). Waves 0-1 each run 3
// v_wmma_f32_16x16x32_f16 with resident B-fragments; all 4 waves share the
// transcendental-heavy cell update. x for the whole block is staged in LDS
// once (64 KB of the WGP's 320 KB).
// ---------------------------------------------------------------------------
__global__ __launch_bounds__(128)
void lstm_wmma_kernel(const float* __restrict__ x,     // [B,C,T] (= [NSEQ,T])
                      const float* __restrict__ Wih,   // [88,1]
                      const float* __restrict__ Whh,   // [88,22]
                      const float* __restrict__ bih,   // [88]
                      const float* __restrict__ bhh,   // [88]
                      _Float16*   __restrict__ hsbuf)  // [B*T, 512] f16
{
    const int tid  = threadIdx.x;
    const int lane = tid & 31;
    const int wave = tid >> 5;
    const int lg   = lane >> 4;      // lane group (0/1)
    const int ln   = lane & 15;
    const int sbase = blockIdx.x * 16;

    __shared__ float    x_all[16][T_LEN];  // staged input, 64 KB
    __shared__ _Float16 h_lds[16][32];     // h state f16, K padded (pad stays 0)
    __shared__ float    c_lds[16][HID];    // c state f32
    __shared__ float    gates[16][96];     // per-step gate tile

    // stage x: the block's 16 rows are contiguous -> fully coalesced
    {
        const float* xrow = x + (size_t)sbase * T_LEN;
        for (int i = tid; i < 16 * T_LEN; i += 128) ((float*)x_all)[i] = xrow[i];
    }
    for (int i = tid; i < 16 * 32;  i += 128) ((_Float16*)h_lds)[i] = (_Float16)0.0f;
    for (int i = tid; i < 16 * HID; i += 128) ((float*)c_lds)[i]    = 0.0f;

    // Waves 0-1: pre-pack B fragments of Whh^T (3 N-tiles each).
    // 16-bit B layout: lane holds column n; group0 elems 0..7 -> K 0..7, 8..15 -> K 16..23,
    // group1 elems 0..7 -> K 8..15, 8..15 -> K 24..31 (K 22..31 zero-padded).
    v16h  bW[3];
    float wih_r[3], bsum_r[3];
    if (wave < 2) {
#pragma unroll
        for (int tt = 0; tt < 3; ++tt) {
            const int n = (wave * 3 + tt) * 16 + ln;
#pragma unroll
            for (int e = 0; e < 16; ++e) {
                const int k = (e < 8) ? (lg * 8 + e) : (16 + lg * 8 + (e - 8));
                float v = (n < NGATE && k < HID) ? Whh[n * HID + k] : 0.0f;
                bW[tt][e] = (_Float16)v;
            }
            wih_r[tt]  = (n < NGATE) ? Wih[n] : 0.0f;
            bsum_r[tt] = (n < NGATE) ? (bih[n] + bhh[n]) : 0.0f;
        }
    }
    __syncthreads();

    for (int t = 0; t < T_LEN; ++t) {
        if (wave < 2) {
            // A fragment from h (ISA 16-bit A 16x32 layout):
            // lane row m = ln; group0: elems 0..7 -> K 0..7, 8..15 -> K 16..23
            //                  group1: elems 0..7 -> K 8..15, 8..15 -> K 24..31
            v16h a;
            const int kb = lg ? 8 : 0;
#pragma unroll
            for (int e = 0; e < 8; ++e) {
                a[e]     = h_lds[ln][kb + e];
                a[e + 8] = h_lds[ln][kb + 16 + e];
            }
#pragma unroll
            for (int tt = 0; tt < 3; ++tt) {
                v8f c;
#pragma unroll
                for (int r = 0; r < 8; ++r) {
                    const int m = r + 8 * lg;               // C layout row
                    c[r] = x_all[m][t] * wih_r[tt] + bsum_r[tt];
                }
                c = __builtin_amdgcn_wmma_f32_16x16x32_f16(
                        false, a, false, bW[tt], (short)0, c, false, false);
#pragma unroll
                for (int r = 0; r < 8; ++r)
                    gates[r + 8 * lg][(wave * 3 + tt) * 16 + ln] = c[r];
            }
        }
        __syncthreads();   // gates visible to all 4 waves

        // LSTM cell: 16 rows x 22 hidden = 352 elems, <=3 per thread
#pragma unroll
        for (int ii = 0; ii < 3; ++ii) {
            const int p = ii * 128 + tid;
            if (p < 16 * HID) {
                const int m = p & 15;
                const int j = p >> 4;              // 0..21
                const float ig = sigmoid_fast(gates[m][j]);
                const float fg = sigmoid_fast(gates[m][HID + j]);
                const float gg = tanh_fast(gates[m][2 * HID + j]);
                const float og = sigmoid_fast(gates[m][3 * HID + j]);
                const float cn = fg * c_lds[m][j] + ig * gg;
                const float hn = og * tanh_fast(cn);
                c_lds[m][j] = cn;
                h_lds[m][j] = (_Float16)hn;
                const int nseq = sbase + m;
                const int b = nseq / CCH, r = nseq % CCH;
                // conv-GEMM-friendly layout: hsbuf[b*T + t][kk = r*22 + j]
                hsbuf[((size_t)b * T_LEN + t) * KPAD + r * HID + j] = (_Float16)hn;
            }
        }
        __syncthreads();   // h/c updated before next step's A build / gate store
    }
}

// ---------------------------------------------------------------------------
// Kernel 2: spatial conv as GEMM + bias + ELU + BatchNorm (fused epilogue).
// out[b,n,t] = sum_kk hs[b,t,kk] * W[kk,n], kk = r*22 + i,
//   W[kk,n] = convw[(n*22 + i)*22 + r]    (OIHW [22,22,22,1])
// One 8-wave block per batch element: weights packed into 32 KB LDS once,
// each wave sweeps t-tiles (16 t x 32 n, 22 valid; K = 512 -> 16 WMMA steps).
// A-fragments are two aligned global_load_b128 per lane per k-step.
// ---------------------------------------------------------------------------
__global__ __launch_bounds__(256)
void conv_wmma_kernel(const _Float16* __restrict__ hsbuf,  // [B*T, 512]
                      const float* __restrict__ convw,
                      const float* __restrict__ convb,
                      const float* __restrict__ gamma,
                      const float* __restrict__ beta,
                      const float* __restrict__ mean,
                      const float* __restrict__ var,
                      float* __restrict__ convout)          // [B,22,T]
{
    const int tid  = threadIdx.x;
    const int lane = tid & 31;
    const int wave = tid >> 5;
    const int lg = lane >> 4, ln = lane & 15;
    const int b  = blockIdx.x;

    __shared__ v16h bw[2][16][32];   // packed weight B-fragments, 32 KB

    // Cooperative weight pack: 1024 fragments across 256 threads.
    for (int f = tid; f < 2 * 16 * 32; f += 256) {
        const int slot = f & 31;
        const int ks   = (f >> 5) & 15;
        const int tn   = f >> 9;
        const int n    = tn * 16 + (slot & 15);
        const int klg  = slot >> 4;
        v16h frag;
#pragma unroll
        for (int e = 0; e < 16; ++e) {
            const int kk = ks * 32 + ((e < 8) ? (klg * 8 + e) : (16 + klg * 8 + (e - 8)));
            float v = 0.0f;
            if (n < HID && kk < KDIM) {
                const int r = kk / HID, i = kk % HID;
                v = convw[(n * HID + i) * HID + r];
            }
            frag[e] = (_Float16)v;
        }
        bw[tn][ks][slot] = frag;
    }
    __syncthreads();

    // per-lane epilogue constants (column n fixed per lane per tn)
    float bias[2], inv[2], mu[2], bet[2];
#pragma unroll
    for (int tn = 0; tn < 2; ++tn) {
        const int n = tn * 16 + ln;
        const bool v = (n < HID);
        bias[tn] = v ? convb[n] : 0.0f;
        inv[tn]  = v ? gamma[n] * rsqrtf(var[n] + 1e-5f) : 0.0f;
        mu[tn]   = v ? mean[n] : 0.0f;
        bet[tn]  = v ? beta[n] : 0.0f;
    }

    // each wave sweeps t-tiles with stride 8 (63 tiles total)
    for (int tt = wave; tt < (T_LEN + 15) / 16; tt += 8) {
        const int t0 = tt * 16;
        int t = t0 + ln;                        // A row handled by this lane
        if (t >= T_LEN) t = T_LEN - 1;          // clamp; stores guarded below
        const v8h* rp = (const v8h*)(hsbuf + ((size_t)b * T_LEN + t) * KPAD);

        v8f acc0 = {}, acc1 = {};
        for (int ks = 0; ks < 16; ++ks) {
            // A elems kb..kb+7 and kb+16..kb+23, kb = ks*32 + lg*8:
            // two aligned 16B chunks of the row
            const v8h lo = rp[ks * 4 + lg];
            const v8h hi = rp[ks * 4 + lg + 2];
            const v16h a = __builtin_shufflevector(
                lo, hi, 0, 1, 2, 3, 4, 5, 6, 7, 8, 9, 10, 11, 12, 13, 14, 15);
            acc0 = __builtin_amdgcn_wmma_f32_16x16x32_f16(
                       false, a, false, bw[0][ks][lane], (short)0, acc0, false, false);
            acc1 = __builtin_amdgcn_wmma_f32_16x16x32_f16(
                       false, a, false, bw[1][ks][lane], (short)0, acc1, false, false);
        }

        // fused epilogue: conv bias -> ELU -> BatchNorm(eval) -> store f32
#pragma unroll
        for (int tn = 0; tn < 2; ++tn) {
            const int n = tn * 16 + ln;
            if (n >= HID) continue;
            const v8f acc = tn ? acc1 : acc0;
#pragma unroll
            for (int r = 0; r < 8; ++r) {
                const int tw = t0 + r + 8 * lg;   // C layout row -> time index
                if (tw < T_LEN) {
                    float v = acc[r] + bias[tn];
                    v = (v > 0.0f) ? v : (__expf(v) - 1.0f);
                    v = (v - mu[tn]) * inv[tn] + bet[tn];
                    convout[((size_t)b * HID + n) * T_LEN + tw] = v;
                }
            }
        }
    }
}

// ---------------------------------------------------------------------------
// Kernel 3: AvgPool2d((1,100)) + Linear(220 -> 4). Negligible cost.
// ---------------------------------------------------------------------------
__global__ __launch_bounds__(256)
void pool_fc_kernel(const float* __restrict__ convout,  // [B,22,T]
                    const float* __restrict__ fcw,      // [4,220]
                    const float* __restrict__ fcb,      // [4]
                    float* __restrict__ out)            // [B,4]
{
    const int b = blockIdx.x;
    const int tid = threadIdx.x;
    __shared__ float pooled[220];

    if (tid < 220) {
        const int k = tid / 10, q = tid % 10;
        const float* p = convout + ((size_t)b * HID + k) * T_LEN + q * 100;
        float s = 0.0f;
        for (int i = 0; i < 100; ++i) s += p[i];
        pooled[tid] = s * 0.01f;
    }
    __syncthreads();

    if (tid < 4) {
        float s = fcb[tid];
        for (int f = 0; f < 220; ++f) s += fcw[tid * 220 + f] * pooled[f];
        out[(size_t)b * 4 + tid] = s;
    }
}

// ---------------------------------------------------------------------------
extern "C" void kernel_launch(void* const* d_in, const int* in_sizes, int n_in,
                              void* d_out, int out_size, void* d_ws, size_t ws_size,
                              hipStream_t stream)
{
    const float* x     = (const float*)d_in[0];
    const float* Wih   = (const float*)d_in[1];
    const float* Whh   = (const float*)d_in[2];
    const float* bih   = (const float*)d_in[3];
    const float* bhh   = (const float*)d_in[4];
    const float* convw = (const float*)d_in[5];
    const float* convb = (const float*)d_in[6];
    const float* gamma = (const float*)d_in[7];
    const float* beta  = (const float*)d_in[8];
    const float* mean  = (const float*)d_in[9];
    const float* var   = (const float*)d_in[10];
    const float* fcw   = (const float*)d_in[11];
    const float* fcb   = (const float*)d_in[12];
    float* out = (float*)d_out;

    // workspace: hs f16 [B*T, 512] (65.5 MB) then conv output f32 [B,22,T]
    _Float16* hsbuf = (_Float16*)d_ws;
    float* convout  = (float*)((char*)d_ws +
                        (size_t)BATCH * T_LEN * KPAD * sizeof(_Float16));

    // zero K-padding of hs rows (disjoint from LSTM's writes -> can precede it)
    pad_zero_kernel<<<(BATCH * T_LEN + 255) / 256, 256, 0, stream>>>(hsbuf);

    lstm_wmma_kernel<<<NSEQ / 16, 128, 0, stream>>>(x, Wih, Whh, bih, bhh, hsbuf);

    conv_wmma_kernel<<<BATCH, 256, 0, stream>>>(
        hsbuf, convw, convb, gamma, beta, mean, var, convout);

    pool_fc_kernel<<<BATCH, 256, 0, stream>>>(convout, fcw, fcb, out);
}
